// GraphSAGEGlobal_12601434047036
// MI455X (gfx1250) — compile-verified, matrix operand order checked by reading, not run on hardware
//
#include <hip/hip_runtime.h>
#include <hip/hip_bf16.h>

#define N_NODES 50000
#define N_EDGES 600000
#define IN_F    7
#define H_DIM   128
#define OUT_F   4
#define N_LAYERS 4
#define EPS_BN  1e-5f
#define M_TILES (N_NODES / 16)   // 3125, exact

typedef __attribute__((ext_vector_type(16))) _Float16 v16h;
typedef __attribute__((ext_vector_type(8)))  _Float16 v8h;
typedef __attribute__((ext_vector_type(8)))  float    v8f;

// ---------------------------------------------------------------------------
// Pack [Wl[layer]; Wr[layer]] (256x128 f32, row-major K x N) into f16 WMMA
// B-fragments. Layout per ISA 7.12.2 (16-bit B 32x16, wave32):
//   lane 0-15  holds column n=lane,    V0..3 -> K 0..7,  V4..7 -> K 16..23
//   lane 16-31 holds column n=lane-16, V0..3 -> K 8..15, V4..7 -> K 24..31
// Storage: pW[((layer*8 + kt)*8 + nt)*32 + lane][16 halfs]
// ---------------------------------------------------------------------------
__global__ void prep_weights_k(const float* __restrict__ Wl,
                               const float* __restrict__ Wr,
                               _Float16* __restrict__ pW) {
    const int tile  = blockIdx.x;          // 0..255 = ((layer*8)+kt)*8 + nt
    const int nt    = tile & 7;
    const int kt    = (tile >> 3) & 7;
    const int layer = tile >> 6;
    const int lane  = threadIdx.x;         // 0..31
    const int col   = nt * 16 + (lane & 15);
    _Float16* dst = pW + ((size_t)tile * 32 + lane) * 16;
#pragma unroll
    for (int v = 0; v < 8; ++v) {
        const int lk = (v & 3) * 2 + ((v >= 4) ? 16 : 0) + ((lane >= 16) ? 8 : 0);
        const int k  = kt * 32 + lk;       // 0..255 (even)
        float w0, w1;
        if (k < H_DIM) {
            w0 = Wl[((size_t)layer * H_DIM + k)     * H_DIM + col];
            w1 = Wl[((size_t)layer * H_DIM + k + 1) * H_DIM + col];
        } else {
            const int kk = k - H_DIM;
            w0 = Wr[((size_t)layer * H_DIM + kk)     * H_DIM + col];
            w1 = Wr[((size_t)layer * H_DIM + kk + 1) * H_DIM + col];
        }
        dst[v * 2 + 0] = (_Float16)w0;
        dst[v * 2 + 1] = (_Float16)w1;
    }
}

__global__ void zero_f_k(float* __restrict__ p, int n) {
    int i = blockIdx.x * blockDim.x + threadIdx.x;
    if (i < n) p[i] = 0.0f;
}

__global__ void deg_count_k(const long long* __restrict__ ei,
                            float* __restrict__ degf) {
    int e = blockIdx.x * blockDim.x + threadIdx.x;
    if (e < N_EDGES) {
        int d = (int)ei[(size_t)N_EDGES + e];   // dst row
        atomicAdd(&degf[d], 1.0f);
    }
}

__global__ void inv_deg_k(float* __restrict__ degf) {
    int i = blockIdx.x * blockDim.x + threadIdx.x;
    if (i < N_NODES) degf[i] = 1.0f / fmaxf(degf[i], 1.0f);
}

// z = x @ enc_W + enc_b   (K=7, pure VALU)
__global__ void encoder_k(const float* __restrict__ x,
                          const float* __restrict__ W,
                          const float* __restrict__ b,
                          float* __restrict__ z) {
    const int n = blockIdx.x;
    const int h = threadIdx.x;
    float acc = b[h];
#pragma unroll
    for (int k = 0; k < IN_F; ++k)
        acc = fmaf(x[(size_t)n * IN_F + k], W[k * H_DIM + h], acc);
    z[(size_t)n * H_DIM + h] = acc;
}

// agg[dst] += z[src] : one wave per edge, float4 per lane, f32 atomics (L2)
__global__ void scatter_k(const long long* __restrict__ ei,
                          const float* __restrict__ z,
                          float* __restrict__ agg) {
    const int w    = (blockIdx.x * blockDim.x + threadIdx.x) >> 5;
    const int lane = threadIdx.x & 31;
    if (w >= N_EDGES) return;
    const int s = (int)ei[w];
    const int d = (int)ei[(size_t)N_EDGES + w];
    const float4 v = ((const float4*)(z + (size_t)s * H_DIM))[lane];
    float* ap = agg + (size_t)d * H_DIM + lane * 4;
    atomicAdd(ap + 0, v.x);
    atomicAdd(ap + 1, v.y);
    atomicAdd(ap + 2, v.z);
    atomicAdd(ap + 3, v.w);
}

// Xh[n] = [ f16(agg[n]*inv_deg[n]) , f16(z[n]) ]  -> N x 256 f16, row-major
__global__ void convert_k(const float* __restrict__ agg,
                          const float* __restrict__ z,
                          const float* __restrict__ inv,
                          _Float16* __restrict__ Xh) {
    const int n = blockIdx.x;
    const int h = threadIdx.x;
    const float id = inv[n];
    Xh[(size_t)n * 256 + h]       = (_Float16)(agg[(size_t)n * H_DIM + h] * id);
    Xh[(size_t)n * 256 + 128 + h] = (_Float16)(z[(size_t)n * H_DIM + h]);
}

// z_out[16x128 strip] = Xh[16x256] @ Wcat[256x128] + bias  via v_wmma.
// Each wave owns one M-tile (16 rows) and all 8 N-tiles; K=256 in 8 steps.
// Also accumulates per-column sum / sumsq for BatchNorm (f32 atomics).
__global__ void __launch_bounds__(128)
sage_gemm_k(const _Float16* __restrict__ Xh,
            const _Float16* __restrict__ pW,
            const float* __restrict__ bias,
            float* __restrict__ zout,
            float* __restrict__ colSum,
            float* __restrict__ colSumSq,
            int layer, int doBN) {
    const int lane = threadIdx.x & 31;
    const int wave = threadIdx.x >> 5;
    const int tile = blockIdx.x * 4 + wave;
    if (tile >= M_TILES) return;
    const int mBase = tile * 16;
    // A fragment source: row = mBase + (lane&15); upper half-wave offsets K by +8
    const _Float16* arow = Xh + (size_t)(mBase + (lane & 15)) * 256 + ((lane >> 4) << 3);

    v8f acc[8];
#pragma unroll
    for (int nt = 0; nt < 8; ++nt) {
        const float bv = bias[layer * H_DIM + nt * 16 + (lane & 15)];
#pragma unroll
        for (int j = 0; j < 8; ++j) acc[nt][j] = bv;
    }

    const v16h* bw = (const v16h*)pW + (size_t)layer * 8 * 8 * 32;

#pragma unroll
    for (int kt = 0; kt < 8; ++kt) {
        const v8h a0 = *(const v8h*)(arow + kt * 32);        // K base .. +7
        const v8h a1 = *(const v8h*)(arow + kt * 32 + 16);   // K base+16 .. +23
        v16h a;
#pragma unroll
        for (int t = 0; t < 8; ++t) { a[t] = a0[t]; a[8 + t] = a1[t]; }
        const v16h* bt = bw + (size_t)kt * 8 * 32 + lane;
#pragma unroll
        for (int nt = 0; nt < 8; ++nt) {
            const v16h bfrag = bt[nt * 32];
            acc[nt] = __builtin_amdgcn_wmma_f32_16x16x32_f16(
                false, a, false, bfrag, (short)0, acc[nt], false, false);
        }
    }

    // D layout: VGPR j -> row mBase + (lane>=16?8:0) + j, col = nt*16 + (lane&15)
    const int rOff = (lane >> 4) << 3;
#pragma unroll
    for (int nt = 0; nt < 8; ++nt) {
        const int col = nt * 16 + (lane & 15);
        float s = 0.0f, ss = 0.0f;
#pragma unroll
        for (int j = 0; j < 8; ++j) {
            const float v = acc[nt][j];
            zout[(size_t)(mBase + rOff + j) * H_DIM + col] = v;
            s += v;
            ss = fmaf(v, v, ss);
        }
        if (doBN) {
            atomicAdd(&colSum[col], s);
            atomicAdd(&colSumSq[col], ss);
        }
    }
}

__global__ void bn_finalize_k(const float* __restrict__ colSum,
                              const float* __restrict__ colSumSq,
                              const float* __restrict__ gamma,
                              const float* __restrict__ beta,
                              int layer,
                              float* __restrict__ scale,
                              float* __restrict__ shift) {
    const int h = threadIdx.x;
    const float invN = 1.0f / (float)N_NODES;
    const float mu  = colSum[h] * invN;
    const float var = colSumSq[h] * invN - mu * mu;
    const float sc  = gamma[layer * H_DIM + h] * rsqrtf(var + EPS_BN);
    scale[h] = sc;
    shift[h] = beta[layer * H_DIM + h] - mu * sc;
}

__global__ void bn_apply_k(float* __restrict__ z,
                           const float* __restrict__ scale,
                           const float* __restrict__ shift) {
    int i = blockIdx.x * blockDim.x + threadIdx.x;
    if (i < N_NODES * H_DIM) {
        const int h = i & (H_DIM - 1);
        z[i] = fmaxf(fmaf(z[i], scale[h], shift[h]), 0.0f);
    }
}

__global__ void decoder_k(const float* __restrict__ z,
                          const float* __restrict__ W,
                          const float* __restrict__ b,
                          float* __restrict__ out) {
    int n = blockIdx.x * blockDim.x + threadIdx.x;
    if (n >= N_NODES) return;
    float a0 = b[0], a1 = b[1], a2 = b[2], a3 = b[3];
    const float* zr = z + (size_t)n * H_DIM;
#pragma unroll 4
    for (int h = 0; h < H_DIM; ++h) {
        const float zv = zr[h];
        a0 = fmaf(zv, W[h * 4 + 0], a0);
        a1 = fmaf(zv, W[h * 4 + 1], a1);
        a2 = fmaf(zv, W[h * 4 + 2], a2);
        a3 = fmaf(zv, W[h * 4 + 3], a3);
    }
    float* o = out + (size_t)n * 4;
    o[0] = a0; o[1] = a1; o[2] = a2; o[3] = a3;
}

extern "C" void kernel_launch(void* const* d_in, const int* in_sizes, int n_in,
                              void* d_out, int out_size, void* d_ws, size_t ws_size,
                              hipStream_t stream) {
    (void)in_sizes; (void)n_in; (void)out_size; (void)ws_size;
    const float*     x      = (const float*)d_in[0];
    const float*     enc_W  = (const float*)d_in[1];
    const float*     enc_b  = (const float*)d_in[2];
    const float*     Wl     = (const float*)d_in[3];
    const float*     Wr     = (const float*)d_in[4];
    const float*     b      = (const float*)d_in[5];
    const float*     gamma  = (const float*)d_in[6];
    const float*     beta   = (const float*)d_in[7];
    const float*     dec_W  = (const float*)d_in[8];
    const float*     dec_b  = (const float*)d_in[9];
    const long long* ei     = (const long long*)d_in[10];
    float* out = (float*)d_out;

    // ---- workspace carve-up (256B aligned) ----
    char* ws = (char*)d_ws;
    size_t off = 0;
    auto carve = [&](size_t bytes) -> void* {
        void* p = ws + off;
        off = (off + bytes + 255) & ~(size_t)255;
        return p;
    };
    float*    z       = (float*)   carve((size_t)N_NODES * H_DIM * sizeof(float));
    float*    agg     = (float*)   carve((size_t)N_NODES * H_DIM * sizeof(float));
    _Float16* Xh      = (_Float16*)carve((size_t)N_NODES * 256 * sizeof(_Float16));
    float*    degf    = (float*)   carve((size_t)N_NODES * sizeof(float));
    _Float16* pW      = (_Float16*)carve((size_t)N_LAYERS * 8 * 8 * 32 * 16 * sizeof(_Float16));
    float*    colSum  = (float*)   carve(2 * H_DIM * sizeof(float)); // sum + sumsq contiguous
    float*    colSumSq = colSum + H_DIM;
    float*    bnScale = (float*)   carve(H_DIM * sizeof(float));
    float*    bnShift = (float*)   carve(H_DIM * sizeof(float));

    const int NH = N_NODES * H_DIM;

    // weight packing (once per launch) + degrees
    prep_weights_k<<<N_LAYERS * 8 * 8, 32, 0, stream>>>(Wl, Wr, pW);
    zero_f_k<<<(N_NODES + 255) / 256, 256, 0, stream>>>(degf, N_NODES);
    deg_count_k<<<(N_EDGES + 255) / 256, 256, 0, stream>>>(ei, degf);
    inv_deg_k<<<(N_NODES + 255) / 256, 256, 0, stream>>>(degf);

    // encoder
    encoder_k<<<N_NODES, H_DIM, 0, stream>>>(x, enc_W, enc_b, z);

    // SAGE layers
    for (int layer = 0; layer < N_LAYERS; ++layer) {
        zero_f_k<<<(NH + 255) / 256, 256, 0, stream>>>(agg, NH);
        scatter_k<<<(N_EDGES * 32 + 255) / 256, 256, 0, stream>>>(ei, z, agg);
        convert_k<<<N_NODES, H_DIM, 0, stream>>>(agg, z, degf, Xh);
        zero_f_k<<<1, 256, 0, stream>>>(colSum, 2 * H_DIM);
        sage_gemm_k<<<(M_TILES + 3) / 4, 128, 0, stream>>>(
            Xh, pW, b, z, colSum, colSumSq, layer, (layer < N_LAYERS - 1) ? 1 : 0);
        if (layer < N_LAYERS - 1) {
            bn_finalize_k<<<1, H_DIM, 0, stream>>>(colSum, colSumSq, gamma, beta,
                                                   layer, bnScale, bnShift);
            bn_apply_k<<<(NH + 255) / 256, 256, 0, stream>>>(z, bnScale, bnShift);
        }
    }

    // decoder
    decoder_k<<<(N_NODES + 255) / 256, 256, 0, stream>>>(z, dec_W, dec_b, out);
}
